// SQHHBlock_68702296867234
// MI455X (gfx1250) — matrix operand order repaired and checked by reading, compiled, not run on hardware
//
#include <hip/hip_runtime.h>
#include <hip/hip_bf16.h>
#include <math.h>

// ---------------------------------------------------------------------------
// SQHH block for MI455X (gfx1250, wave32, WMMA).
// Shapes fixed by the reference: B=8, N=1024, D=512, KA=16, H=8.
// ---------------------------------------------------------------------------
constexpr int kB  = 8;
constexpr int kN  = 1024;
constexpr int kD  = 512;
constexpr int kKA = 16;
constexpr int kH  = 8;
constexpr int kQD = kD / 4;   // 128
constexpr int kHD = kD / kH;  // 64
constexpr int kBN = kB * kN;  // 8192

typedef _Float16 half16v __attribute__((ext_vector_type(16)));
typedef float    float8v __attribute__((ext_vector_type(8)));

struct __attribute__((aligned(16))) H8 { _Float16 h[8]; };
union AFrag { half16v v; H8 h8[2]; };

// ---------------------------------------------------------------------------
// 1) Expand quaternion weight [4, QD, QD] (f32) into Hamilton block matrix
//    Wt[512][512] (f16), row-major over [n_out][k_in], ready for B-operand.
// ---------------------------------------------------------------------------
__global__ __launch_bounds__(256)
void expand_w_kernel(const float* __restrict__ w, _Float16* __restrict__ Wt) {
  int idx = blockIdx.x * 256 + threadIdx.x;     // 512*512 elements
  if (idx >= kD * kD) return;
  int row = idx >> 9;          // n_out
  int col = idx & (kD - 1);    // k_in
  int a = row >> 7, o  = row & (kQD - 1);
  int c = col >> 7, ii = col & (kQD - 1);
  // Hamilton block table: rows [r,-i,-j,-k | i,r,-k,j | j,k,r,-i | k,-j,i,r]
  const int   comp[4][4] = {{0,1,2,3},{1,0,3,2},{2,3,0,1},{3,2,1,0}};
  const float sgn [4][4] = {{1.f,-1.f,-1.f,-1.f},
                            {1.f, 1.f,-1.f, 1.f},
                            {1.f, 1.f, 1.f,-1.f},
                            {1.f,-1.f, 1.f, 1.f}};
  float v = sgn[a][c] * w[(size_t)comp[a][c] * kQD * kQD + (size_t)o * kQD + ii];
  Wt[(size_t)row * kD + col] = (_Float16)v;
}

// ---------------------------------------------------------------------------
// 2) Spike quaternion rotation: q_rot = (c,0,0,s) ⊗ q, plus per-lane
//    quaternion norms u[b,n,t] = |q_rot_t|^2 for the anchor distance.
// ---------------------------------------------------------------------------
__global__ __launch_bounds__(256)
void rotate_kernel(const float* __restrict__ q, const float* __restrict__ spike,
                   const float* __restrict__ theta_logit,
                   float* __restrict__ qrotF, _Float16* __restrict__ qrotH,
                   float* __restrict__ u) {
  int idx = blockIdx.x * 256 + threadIdx.x;     // B*N*QD threads
  if (idx >= kBN * kQD) return;
  int t  = idx & (kQD - 1);
  int bn = idx >> 7;
  float tl   = theta_logit[0];
  float tmax = 1.5707963267948966f / (1.0f + __expf(-tl));
  float th   = tmax * spike[bn];
  float c = __cosf(th), s = __sinf(th);
  const float* qp = q + (size_t)bn * kD;
  float qr = qp[t], qi = qp[kQD + t], qj = qp[2 * kQD + t], qk = qp[3 * kQD + t];
  float orr = c * qr - s * qk;
  float oi  = c * qi - s * qj;
  float oj  = c * qj + s * qi;
  float ok  = c * qk + s * qr;
  float* of = qrotF + (size_t)bn * kD;
  of[t] = orr; of[kQD + t] = oi; of[2 * kQD + t] = oj; of[3 * kQD + t] = ok;
  _Float16* oh = qrotH + (size_t)bn * kD;
  oh[t] = (_Float16)orr; oh[kQD + t] = (_Float16)oi;
  oh[2 * kQD + t] = (_Float16)oj; oh[3 * kQD + t] = (_Float16)ok;
  u[(size_t)bn * kQD + t] = orr * orr + oi * oi + oj * oj + ok * ok;
}

// ---------------------------------------------------------------------------
// 3) WMMA GEMM: out[M, Nout] = A[M, K] @ Wt[Nout, K]^T + bias.
//    A, Wt f16 row-major; f32 accumulation via v_wmma_f32_16x16x32_f16.
//    Block (32,4): 4 waves, each owns a 32x64 tile (8 accumulators); the
//    K loop is double-buffered (ping-pong fragments, K-step 64) so loads of
//    the next step are in flight while the current WMMA group executes.
//    Grid: (M/128, Nout/64). M and Nout must be multiples of 128 / 64.
// ---------------------------------------------------------------------------
struct GemmFrags { AFrag a0, a1, b[4]; };

__global__ __launch_bounds__(128)
void gemm_wmma_kernel(const _Float16* __restrict__ A,
                      const _Float16* __restrict__ Wt,
                      const float* __restrict__ bias,
                      float* __restrict__ outF,
                      _Float16* __restrict__ outH,
                      int M, int K, int Nout) {
  const int lane    = threadIdx.x;           // 0..31
  const int wave    = threadIdx.y;           // 0..3
  const int laneLow = lane & 15;
  const int laneHi  = lane >> 4;             // 0 or 1
  const int m0 = blockIdx.x * 128 + wave * 32;
  const int n0 = blockIdx.y * 64;

  float8v acc[2][4] = {};

  // A layout (16-bit 16x32): lanes 0-15 -> K{0..7,16..23}, lanes 16-31 -> +8.
  const _Float16* aBase0 = A + (size_t)(m0 + laneLow) * K + laneHi * 8;
  const _Float16* aBase1 = aBase0 + (size_t)16 * K;
  // B layout (32x16): lane holds column n = lane&15, same K order as A
  // (lanes 0-15: K{0..7,16..23}, lanes 16-31: +8).
  const _Float16* bBase[4];
#pragma unroll
  for (int nt = 0; nt < 4; ++nt)
    bBase[nt] = Wt + (size_t)(n0 + nt * 16 + laneLow) * K + laneHi * 8;

  auto loadFrags = [&](GemmFrags& f, int k0) {
    f.a0.h8[0] = *(const H8*)(aBase0 + k0);
    f.a0.h8[1] = *(const H8*)(aBase0 + k0 + 16);
    f.a1.h8[0] = *(const H8*)(aBase1 + k0);
    f.a1.h8[1] = *(const H8*)(aBase1 + k0 + 16);
#pragma unroll
    for (int nt = 0; nt < 4; ++nt) {
      f.b[nt].h8[0] = *(const H8*)(bBase[nt] + k0);
      f.b[nt].h8[1] = *(const H8*)(bBase[nt] + k0 + 16);
    }
  };
  auto compute = [&](GemmFrags& f) {
#pragma unroll
    for (int nt = 0; nt < 4; ++nt) {
      acc[0][nt] = __builtin_amdgcn_wmma_f32_16x16x32_f16(
          false, f.a0.v, false, f.b[nt].v, (short)0, acc[0][nt], false, false);
      acc[1][nt] = __builtin_amdgcn_wmma_f32_16x16x32_f16(
          false, f.a1.v, false, f.b[nt].v, (short)0, acc[1][nt], false, false);
    }
  };

  GemmFrags cur, nxt;
  loadFrags(cur, 0);
  for (int k0 = 0; k0 < K; k0 += 64) {         // K multiple of 64
    loadFrags(nxt, k0 + 32);
    compute(cur);
    if (k0 + 64 < K) loadFrags(cur, k0 + 64);
    compute(nxt);
  }

#pragma unroll
  for (int nt = 0; nt < 4; ++nt) {
    const int ncol = n0 + nt * 16 + laneLow;
    const float bv = bias ? bias[ncol] : 0.0f;
#pragma unroll
    for (int mt = 0; mt < 2; ++mt) {
#pragma unroll
      for (int r = 0; r < 8; ++r) {
        const int mrow = m0 + mt * 16 + r + laneHi * 8;  // C/D: M = r + 8*laneHi
        float v = acc[mt][nt][r] + bv;
        if (outF) outF[(size_t)mrow * Nout + ncol] = v;
        if (outH) outH[(size_t)mrow * Nout + ncol] = (_Float16)v;
      }
    }
  }
}

// ---------------------------------------------------------------------------
// 4) Anchor prep: vanch[k,t] = |anchor quat|^2 ; sigsq[k] = max(exp(ls)^2,1e-6)
// ---------------------------------------------------------------------------
__global__ __launch_bounds__(128)
void anchor_prep_kernel(const float* __restrict__ anchors,
                        const float* __restrict__ log_sigma,
                        float* __restrict__ vanch, float* __restrict__ sigsq) {
  int idx = blockIdx.x * 128 + threadIdx.x;      // KA*QD = 2048
  if (idx >= kKA * kQD) return;
  int k = idx >> 7, t = idx & (kQD - 1);
  const float* ap = anchors + (size_t)k * kD;
  float r = ap[t], i = ap[kQD + t], j = ap[2 * kQD + t], kk = ap[3 * kQD + t];
  vanch[idx] = r * r + i * i + j * j + kk * kk;
  if (t == 0) {
    float s = __expf(log_sigma[k]);
    sigsq[k] = fmaxf(s * s, 1e-6f);
  }
}

// ---------------------------------------------------------------------------
// 5) Incidence: dist[b,k,n] = u[b,n,:]·vanch[k,:]  (quaternion-norm identity)
//    incid = exp(-dist/sigsq[k]) * mask[b,n]
// ---------------------------------------------------------------------------
__global__ __launch_bounds__(256)
void incid_kernel(const float* __restrict__ u, const float* __restrict__ vanch,
                  const float* __restrict__ sigsq, const float* __restrict__ mask,
                  float* __restrict__ incid) {
  int idx = blockIdx.x * 256 + threadIdx.x;      // B*KA*N = 131072
  if (idx >= kB * kKA * kN) return;
  int n = idx & (kN - 1);
  int k = (idx >> 10) & (kKA - 1);
  int b = idx >> 14;
  const float* up = u + ((size_t)b * kN + n) * kQD;
  const float* vp = vanch + (size_t)k * kQD;
  float d = 0.0f;
#pragma unroll 4
  for (int t = 0; t < kQD; ++t) d += up[t] * vp[t];
  incid[idx] = __expf(-d / sigsq[k]) * mask[(size_t)b * kN + n];
}

// ---------------------------------------------------------------------------
// 6) Anchor aggregation: h0[b,k,d] = (Σ_n incid·q_rot[b,n,d]) / max(Σ incid,1e-6)
// ---------------------------------------------------------------------------
__global__ __launch_bounds__(256)
void aggregate_kernel(const float* __restrict__ incid,
                      const float* __restrict__ qrotF,
                      _Float16* __restrict__ h0H) {
  int bk = blockIdx.x;                           // B*KA blocks
  int b  = bk / kKA;
  const float* ip = incid + (size_t)bk * kN;
  __shared__ float red[256];
  float s = 0.0f;
  for (int n = threadIdx.x; n < kN; n += 256) s += ip[n];
  red[threadIdx.x] = s;
  __syncthreads();
  for (int st = 128; st > 0; st >>= 1) {
    if ((int)threadIdx.x < st) red[threadIdx.x] += red[threadIdx.x + st];
    __syncthreads();
  }
  float invDen = 1.0f / fmaxf(red[0], 1e-6f);
  for (int d0 = threadIdx.x; d0 < kD; d0 += 256) {
    const float* qp = qrotF + (size_t)b * kN * kD + d0;
    float acc = 0.0f;
    for (int n = 0; n < kN; ++n) acc += ip[n] * qp[(size_t)n * kD];
    h0H[(size_t)bk * kD + d0] = (_Float16)(acc * invDen);
  }
}

// ---------------------------------------------------------------------------
// 7) Attention over KA=16 keys, H=8 heads. One 256-thread block per (b,n).
// ---------------------------------------------------------------------------
__global__ __launch_bounds__(256)
void attn_kernel(const float* __restrict__ Qh, const float* __restrict__ Kh,
                 const float* __restrict__ Vh, _Float16* __restrict__ outH) {
  int bn = blockIdx.x;
  int b  = bn >> 10;
  const float* qrow  = Qh + (size_t)bn * kD;
  const float* kbase = Kh + (size_t)b * kKA * kD;
  const float* vbase = Vh + (size_t)b * kKA * kD;
  __shared__ float part[256];
  __shared__ float sAttn[kH * kKA];              // 128
  int t = threadIdx.x;
  {  // partial dot products: thread = (head hh, key k, half)
    int hh = t >> 5, k = (t >> 1) & 15, half = t & 1;
    const float* qp = qrow + hh * kHD + half * 32;
    const float* kp = kbase + (size_t)k * kD + hh * kHD + half * 32;
    float s = 0.0f;
#pragma unroll 4
    for (int i = 0; i < 32; ++i) s += qp[i] * kp[i];
    part[t] = s;
  }
  __syncthreads();
  if (t < 128) {                                 // scores: scale = 1/sqrt(64)
    int hh = t >> 4, k = t & 15;
    sAttn[t] = (part[hh * 32 + k * 2] + part[hh * 32 + k * 2 + 1]) * 0.125f;
  }
  __syncthreads();
  if (t < kH) {                                  // softmax per head (16 keys)
    float mx = -1e30f;
    for (int k = 0; k < kKA; ++k) mx = fmaxf(mx, sAttn[t * kKA + k]);
    float sum = 0.0f;
    for (int k = 0; k < kKA; ++k) {
      float e = __expf(sAttn[t * kKA + k] - mx);
      sAttn[t * kKA + k] = e;
      sum += e;
    }
    float inv = 1.0f / sum;
    for (int k = 0; k < kKA; ++k) sAttn[t * kKA + k] *= inv;
  }
  __syncthreads();
  for (int d0 = t; d0 < kD; d0 += 256) {
    int hh = d0 >> 6;
    float acc = 0.0f;
#pragma unroll
    for (int k = 0; k < kKA; ++k)
      acc += sAttn[hh * kKA + k] * vbase[(size_t)k * kD + d0];
    outH[(size_t)bn * kD + d0] = (_Float16)acc;
  }
}

// ---------------------------------------------------------------------------
// 8) Residual + per-component quaternion LayerNorm. 128 threads per (b,n).
// ---------------------------------------------------------------------------
__global__ __launch_bounds__(128)
void final_ln_kernel(const float* __restrict__ q, const float* __restrict__ msgp,
                     const float* __restrict__ msga,
                     const float* __restrict__ gamma, const float* __restrict__ beta,
                     float* __restrict__ out) {
  int bn = blockIdx.x;
  int t  = threadIdx.x;                          // 0..127
  size_t base = (size_t)bn * kD;
  float x[4];
#pragma unroll
  for (int c = 0; c < 4; ++c) {
    size_t o = base + (size_t)c * kQD + t;
    x[c] = q[o] + msgp[o] + msga[o];
  }
  __shared__ float s1[4][128];
  __shared__ float s2[4][128];
#pragma unroll
  for (int c = 0; c < 4; ++c) { s1[c][t] = x[c]; s2[c][t] = x[c] * x[c]; }
  __syncthreads();
  for (int st = 64; st > 0; st >>= 1) {
    if (t < st)
#pragma unroll
      for (int c = 0; c < 4; ++c) {
        s1[c][t] += s1[c][t + st];
        s2[c][t] += s2[c][t + st];
      }
    __syncthreads();
  }
#pragma unroll
  for (int c = 0; c < 4; ++c) {
    float m = s1[c][0] * (1.0f / 128.0f);
    float v = s2[c][0] * (1.0f / 128.0f) - m * m;
    float y = (x[c] - m) * rsqrtf(v + 1e-5f) * gamma[c * kQD + t] + beta[c * kQD + t];
    out[base + (size_t)c * kQD + t] = y;
  }
}

// ---------------------------------------------------------------------------
// Host orchestration
// ---------------------------------------------------------------------------
extern "C" void kernel_launch(void* const* d_in, const int* in_sizes, int n_in,
                              void* d_out, int out_size, void* d_ws, size_t ws_size,
                              hipStream_t stream) {
  const float* q           = (const float*)d_in[0];
  const float* spike       = (const float*)d_in[1];
  const float* mask        = (const float*)d_in[2];
  const float* theta_logit = (const float*)d_in[3];
  const float* w_prim = (const float*)d_in[4];  const float* b_prim = (const float*)d_in[5];
  const float* anchors = (const float*)d_in[6]; const float* log_sigma = (const float*)d_in[7];
  const float* w_aggr = (const float*)d_in[8];  const float* b_aggr = (const float*)d_in[9];
  const float* w_q  = (const float*)d_in[10];   const float* b_q = (const float*)d_in[11];
  const float* w_k  = (const float*)d_in[12];   const float* b_k = (const float*)d_in[13];
  const float* w_v  = (const float*)d_in[14];   const float* b_v = (const float*)d_in[15];
  const float* w_o  = (const float*)d_in[16];   const float* b_o = (const float*)d_in[17];
  const float* ln_gamma = (const float*)d_in[18];
  const float* ln_beta  = (const float*)d_in[19];
  float* out = (float*)d_out;

  char* ws = (char*)d_ws;
  size_t off = 0;
  auto alloc = [&](size_t bytes) -> void* {
    void* p = ws + off;
    off += (bytes + 255) & ~(size_t)255;
    return p;
  };

  float*    qrotF  = (float*)   alloc((size_t)kBN * kD * 4);      // 16 MB
  _Float16* qrotH  = (_Float16*)alloc((size_t)kBN * kD * 2);      //  8 MB
  float*    uNorm  = (float*)   alloc((size_t)kBN * kQD * 4);     //  4 MB
  float*    msgpF  = (float*)   alloc((size_t)kBN * kD * 4);      // 16 MB
  float*    QhF    = (float*)   alloc((size_t)kBN * kD * 4);      // 16 MB (reused for msg_a)
  _Float16* attnH  = (_Float16*)alloc((size_t)kBN * kD * 2);      //  8 MB
  _Float16* Wexp[6];
  for (int i = 0; i < 6; ++i) Wexp[i] = (_Float16*)alloc((size_t)kD * kD * 2);
  _Float16* h0H    = (_Float16*)alloc((size_t)kB * kKA * kD * 2);
  _Float16* h1H    = (_Float16*)alloc((size_t)kB * kKA * kD * 2);
  float*    KhF    = (float*)   alloc((size_t)kB * kKA * kD * 4);
  float*    VhF    = (float*)   alloc((size_t)kB * kKA * kD * 4);
  float*    incid  = (float*)   alloc((size_t)kB * kKA * kN * 4);
  float*    vanch  = (float*)   alloc((size_t)kKA * kQD * 4);
  float*    sigsq  = (float*)   alloc((size_t)kKA * 4);
  float*    msgaF  = QhF;   // attention consumes QhF before w_o GEMM writes it

  // 1) Expand the 6 quaternion weights into Hamilton f16 matrices.
  const float* wsrc[6] = {w_prim, w_aggr, w_q, w_k, w_v, w_o};
  {
    dim3 g((kD * kD + 255) / 256), bl(256);
    for (int i = 0; i < 6; ++i)
      expand_w_kernel<<<g, bl, 0, stream>>>(wsrc[i], Wexp[i]);
  }

  // 2) Spike rotation (+ quaternion norms).
  rotate_kernel<<<(kBN * kQD + 255) / 256, 256, 0, stream>>>(
      q, spike, theta_logit, qrotF, qrotH, uNorm);

  // 3) msg_p = q_rot @ Wprim^T + b_prim   (WMMA)
  {
    dim3 g(kBN / 128, kD / 64), bl(32, 4);
    gemm_wmma_kernel<<<g, bl, 0, stream>>>(qrotH, Wexp[0], b_prim,
                                           msgpF, (_Float16*)nullptr, kBN, kD, kD);
    // 4) Qh = q_rot @ Wq^T + b_q          (WMMA)
    gemm_wmma_kernel<<<g, bl, 0, stream>>>(qrotH, Wexp[2], b_q,
                                           QhF, (_Float16*)nullptr, kBN, kD, kD);
  }

  // 5) Anchor incidence and aggregation.
  anchor_prep_kernel<<<(kKA * kQD + 127) / 128, 128, 0, stream>>>(
      anchors, log_sigma, vanch, sigsq);
  incid_kernel<<<(kB * kKA * kN + 255) / 256, 256, 0, stream>>>(
      uNorm, vanch, sigsq, mask, incid);
  aggregate_kernel<<<kB * kKA, 256, 0, stream>>>(incid, qrotF, h0H);

  // 6) h = h0 @ Waggr^T + b_aggr ; then Kh, Vh.   (WMMA, M = 128)
  {
    dim3 g((kB * kKA) / 128, kD / 64), bl(32, 4);
    gemm_wmma_kernel<<<g, bl, 0, stream>>>(h0H, Wexp[1], b_aggr,
                                           (float*)nullptr, h1H, kB * kKA, kD, kD);
    gemm_wmma_kernel<<<g, bl, 0, stream>>>(h1H, Wexp[3], b_k,
                                           KhF, (_Float16*)nullptr, kB * kKA, kD, kD);
    gemm_wmma_kernel<<<g, bl, 0, stream>>>(h1H, Wexp[4], b_v,
                                           VhF, (_Float16*)nullptr, kB * kKA, kD, kD);
  }

  // 7) Multi-head attention (KA=16 keys).
  attn_kernel<<<kBN, 256, 0, stream>>>(QhF, KhF, VhF, attnH);

  // 8) msg_a = attn_out @ Wo^T + b_o      (WMMA; reuses QhF storage)
  {
    dim3 g(kBN / 128, kD / 64), bl(32, 4);
    gemm_wmma_kernel<<<g, bl, 0, stream>>>(attnH, Wexp[5], b_o,
                                           msgaF, (_Float16*)nullptr, kBN, kD, kD);
  }

  // 9) Residual + quaternion LayerNorm.
  final_ln_kernel<<<kBN, 128, 0, stream>>>(q, msgpF, msgaF, ln_gamma, ln_beta, out);
}
